// PhaseBindingRecall_59974923321459
// MI455X (gfx1250) — matrix operand order repaired and checked by reading, not compiled, not run on hardware
//
#include <hip/hip_runtime.h>
#include <math.h>
#include <stdint.h>

typedef __bf16 bf16_t;
typedef __attribute__((ext_vector_type(16))) __bf16 v16bf;
typedef __attribute__((ext_vector_type(8)))  __bf16 v8bf;
typedef __attribute__((ext_vector_type(8)))  float  v8f;

#define B_      16
#define L_      1024
#define VOCAB_  1024
#define D_      256
#define K_      64
#define NP_     511           // N_PAIRS = (L-2)/2
#define NPP_    512           // padded N_PAIRS (pad entry zeroed)
#define M_      (B_ * NP_)    // 8176 rows (multiple of 16)
#define PI_F    3.14159265358979323846f

__device__ __forceinline__ bf16_t f2bf(float x) { return (bf16_t)x; }

__device__ __forceinline__ float softplus_f(float x) {
  return (x > 20.f) ? x : log1pf(expf(x));
}

__device__ __forceinline__ v8bf ld8(const bf16_t* p) {
  return *(const v8bf*)p;     // 16B aligned -> global_load_b128 / ds_load_b128
}
__device__ __forceinline__ v16bf cat8(v8bf lo, v8bf hi) {
  return __builtin_shufflevector(lo, hi, 0,1,2,3,4,5,6,7,8,9,10,11,12,13,14,15);
}

// CDNA5 async global -> LDS copy (ASYNCcnt-tracked), 16 bytes per op.
// GV mode: vdst = LDS byte address, vaddr = 64-bit global address.
__device__ __forceinline__ void async_g2l_b128(uint32_t lds_off, const void* gptr) {
  asm volatile("global_load_async_to_lds_b128 %0, %1, off"
               :: "v"(lds_off), "v"((uint64_t)(uintptr_t)gptr)
               : "memory");
}
__device__ __forceinline__ void wait_async0() {
  asm volatile("s_wait_asynccnt 0x0" ::: "memory");
}

// ---------------------------------------------------------------------------
// 1) argmax over last dim of x: one 256-thread block per (b,l) row of 1025.
// ---------------------------------------------------------------------------
__global__ void argmax_kernel(const float* __restrict__ x, int* __restrict__ idx) {
  __shared__ float sv[256];
  __shared__ int   si[256];
  const int row = blockIdx.x;                   // b*L + l
  const float* p = x + (size_t)row * (VOCAB_ + 1);
  const int t = threadIdx.x;
  float best = -INFINITY; int bi = 0;
  for (int i = t; i < VOCAB_ + 1; i += 256) {
    float v = p[i];
    if (v > best) { best = v; bi = i; }
  }
  sv[t] = best; si[t] = bi;
  __syncthreads();
  for (int s = 128; s > 0; s >>= 1) {
    if (t < s) {
      if (sv[t + s] > sv[t] || (sv[t + s] == sv[t] && si[t + s] < si[t])) {
        sv[t] = sv[t + s]; si[t] = si[t + s];
      }
    }
    __syncthreads();
  }
  if (t == 0) idx[row] = si[0];
}

// ---------------------------------------------------------------------------
// 2) gather embed rows for keys (even pos) and values (odd pos), cvt -> bf16
// ---------------------------------------------------------------------------
__global__ void gather_kernel(const float* __restrict__ embed, const int* __restrict__ idx,
                              bf16_t* __restrict__ keyh, bf16_t* __restrict__ valh) {
  const int m = blockIdx.x;            // 0..M_-1
  const int b = m / NP_, n = m % NP_;
  const int t = threadIdx.x;           // 0..255 = d
  const int ik = idx[b * L_ + 2 * n];
  const int iv = idx[b * L_ + 2 * n + 1];
  keyh[(size_t)m * D_ + t] = f2bf(embed[(size_t)ik * D_ + t]);
  valh[(size_t)m * D_ + t] = f2bf(embed[(size_t)iv * D_ + t]);
}

// f32 weights [K][N] -> bf16 transposed [N][K] (so B-fragments load contiguously)
__global__ void cvtT_kernel(const float* __restrict__ src, bf16_t* __restrict__ dst,
                            int Krows, int Ncols) {
  int i = blockIdx.x * blockDim.x + threadIdx.x;
  if (i >= Krows * Ncols) return;
  int k = i / Ncols, n = i % Ncols;
  dst[(size_t)n * Krows + k] = f2bf(src[i]);
}

// zero the padded n=511 entry of krT / kiT / Vt rows
__global__ void padzero_kernel(bf16_t* __restrict__ krT, bf16_t* __restrict__ kiT,
                               bf16_t* __restrict__ Vt) {
  int i = blockIdx.x * blockDim.x + threadIdx.x;
  const bf16_t z = (bf16_t)0.f;
  if (i < B_ * K_) { krT[(size_t)i * NPP_ + NP_] = z; kiT[(size_t)i * NPP_ + NP_] = z; }
  if (i < B_ * D_) { Vt[(size_t)i * NPP_ + NP_] = z; }
}

// ---------------------------------------------------------------------------
// 3) WMMA GEMM: out = act(A[M][KD] @ W[KD][N] + bias[N]); W given transposed
//    as Wt[N][KD]. 4 waves/block share the A tile, which is async-staged into
//    LDS once (contiguous 16*KD bf16 block, ASYNCcnt-tracked copy).
//    ACT: 0 = none -> f32 [M][N]
//         1 = tanh -> bf16 [M][N]
//         3 = none -> bf16 transposed+padded Vt[(b*D + n_col)*NPP + n_row]
// ---------------------------------------------------------------------------
template <int ACT, int KD>
__global__ void gemm_wmma_kernel(const bf16_t* __restrict__ A, const bf16_t* __restrict__ Wt,
                                 const float* __restrict__ bias, void* __restrict__ out,
                                 int N) {
  __shared__ bf16_t shA[16 * KD];      // 8KB for KD=256
  const int tid  = threadIdx.x;        // 0..127
  const int wave = tid >> 5;
  const int lane = tid & 31;
  const int m0   = blockIdx.x * 16;
  const int n0   = (blockIdx.y * 4 + wave) * 16;
  const int half = lane >> 4;
  const int l16  = lane & 15;
  const int akb  = half * 8;
  const int bcol = n0 + l16;
  const int bkb  = half * 8;

  // async-stage A tile: rows m0..m0+15 are one contiguous 16*KD*2 byte block
  {
    const char* gsrc = (const char*)(A + (size_t)m0 * KD);
    const uint32_t bytes = 16u * KD * (uint32_t)sizeof(bf16_t);
#pragma unroll
    for (uint32_t off = (uint32_t)tid * 16u; off < bytes; off += 128u * 16u) {
      async_g2l_b128((uint32_t)(uintptr_t)((char*)shA + off), gsrc + off);
    }
    wait_async0();
  }
  __syncthreads();

  const bf16_t* Ap = shA + (size_t)l16 * KD;       // LDS reads (ds_load_b128)
  const bf16_t* Wp = Wt  + (size_t)bcol * KD;      // global b128 reads

  v8f acc = {};
#pragma unroll
  for (int k0 = 0; k0 < KD; k0 += 32) {
    v16bf a  = cat8(ld8(Ap + k0 + akb),  ld8(Ap + k0 + 16 + akb));
    v16bf bf = cat8(ld8(Wp + k0 + bkb),  ld8(Wp + k0 + 16 + bkb));
    acc = __builtin_amdgcn_wmma_f32_16x16x32_bf16(false, a, false, bf,
                                                  (short)0, acc, false, false);
  }
  const float bv = bias[bcol];
#pragma unroll
  for (int r = 0; r < 8; ++r) {
    const int mrow = m0 + r + half * 8;
    float v = acc[r] + bv;
    if (ACT == 1) v = tanhf(v);
    if (ACT == 0) {
      ((float*)out)[(size_t)mrow * N + bcol] = v;
    } else if (ACT == 1) {
      ((bf16_t*)out)[(size_t)mrow * N + bcol] = f2bf(v);
    } else {  // ACT == 3: transposed + padded V
      const int bb = mrow / NP_, nn = mrow % NP_;
      ((bf16_t*)out)[((size_t)bb * D_ + bcol) * NPP_ + nn] = f2bf(v);
    }
  }
}

// ---------------------------------------------------------------------------
// 4) elementwise kr/ki, written transposed+padded: krT[(b*K + k)*NPP + n]
// ---------------------------------------------------------------------------
__global__ void krki_kernel(const float* __restrict__ phasepre, const float* __restrict__ amppre,
                            bf16_t* __restrict__ krT, bf16_t* __restrict__ kiT) {
  int i = blockIdx.x * blockDim.x + threadIdx.x;
  if (i >= M_ * K_) return;
  const int m = i >> 6;          // K_ = 64
  const int k = i & 63;
  const int b = m / NP_, n = m % NP_;
  float phase = tanhf(phasepre[i]) * PI_F;
  float amp = softplus_f(amppre[i]) + 0.1f;
  float s, c;
  sincosf(phase, &s, &c);
  const size_t o = ((size_t)b * K_ + k) * NPP_ + n;
  krT[o] = f2bf(amp * c);
  kiT[o] = f2bf(amp * s);
}

// ---------------------------------------------------------------------------
// 5) binding: mem_r[b][k][d] = sum_n kr[b][n][k] * V[b][n][d] (and mem_i)
//    A = krT/kiT tiles (16 x 512, contiguous 16KB each) async-staged into LDS
//    and shared by the 4 waves; B = Vt rows direct from global (b128).
//    Two WMMAs per step share the V fragment.
// ---------------------------------------------------------------------------
__global__ void bind_wmma_kernel(const bf16_t* __restrict__ krT, const bf16_t* __restrict__ kiT,
                                 const bf16_t* __restrict__ Vt,
                                 float* __restrict__ mem_r, float* __restrict__ mem_i) {
  __shared__ bf16_t shR[16 * NPP_];    // 16KB
  __shared__ bf16_t shI[16 * NPP_];    // 16KB
  const int b    = blockIdx.x;
  const int k0   = blockIdx.y * 16;              // K_/16 = 4
  const int tid  = threadIdx.x;
  const int wave = tid >> 5;
  const int d0   = (blockIdx.z * 4 + wave) * 16; // D_/64 = 4 groups
  const int lane = tid & 31;
  const int half = lane >> 4;
  const int l16  = lane & 15;
  const int anb  = half * 8;
  const int bcol = d0 + l16;                     // d index
  const int bnb  = half * 8;

  // async-stage krT/kiT tiles: rows (b*K_+k0)..(+15), contiguous 16*NPP_*2 B
  {
    const char* gr = (const char*)(krT + ((size_t)b * K_ + k0) * NPP_);
    const char* gi = (const char*)(kiT + ((size_t)b * K_ + k0) * NPP_);
    const uint32_t bytes = 16u * NPP_ * (uint32_t)sizeof(bf16_t);
#pragma unroll
    for (uint32_t off = (uint32_t)tid * 16u; off < bytes; off += 128u * 16u) {
      async_g2l_b128((uint32_t)(uintptr_t)((char*)shR + off), gr + off);
      async_g2l_b128((uint32_t)(uintptr_t)((char*)shI + off), gi + off);
    }
    wait_async0();
  }
  __syncthreads();

  const bf16_t* Arp = shR + (size_t)l16 * NPP_;  // LDS reads
  const bf16_t* Aip = shI + (size_t)l16 * NPP_;
  const bf16_t* Bp  = Vt  + ((size_t)b * D_ + bcol) * NPP_;

  v8f accr = {}, acci = {};
#pragma unroll
  for (int n0 = 0; n0 < NPP_; n0 += 32) {        // 16 uniform steps
    v16bf ar = cat8(ld8(Arp + n0 + anb), ld8(Arp + n0 + 16 + anb));
    v16bf ai = cat8(ld8(Aip + n0 + anb), ld8(Aip + n0 + 16 + anb));
    v16bf bv = cat8(ld8(Bp  + n0 + bnb), ld8(Bp  + n0 + 16 + bnb));
    accr = __builtin_amdgcn_wmma_f32_16x16x32_bf16(false, ar, false, bv,
                                                   (short)0, accr, false, false);
    acci = __builtin_amdgcn_wmma_f32_16x16x32_bf16(false, ai, false, bv,
                                                   (short)0, acci, false, false);
  }
#pragma unroll
  for (int r = 0; r < 8; ++r) {
    const int krow = k0 + r + half * 8;
    mem_r[((size_t)b * K_ + krow) * D_ + bcol] = accr[r];
    mem_i[((size_t)b * K_ + krow) * D_ + bcol] = acci[r];
  }
}

// ---------------------------------------------------------------------------
// 6) head: query phase-encode, retrieval, layernorm, two small GEMMs -> logits
// ---------------------------------------------------------------------------
__global__ void head_kernel(const int* __restrict__ idx, const float* __restrict__ embed,
                            const float* __restrict__ qp_w1, const float* __restrict__ qp_b1,
                            const float* __restrict__ qp_w2, const float* __restrict__ qp_b2,
                            const float* __restrict__ qp_aw, const float* __restrict__ qp_ab,
                            const float* __restrict__ mem_r, const float* __restrict__ mem_i,
                            const float* __restrict__ ln_g, const float* __restrict__ ln_b,
                            const float* __restrict__ ow, const float* __restrict__ ob,
                            const float* __restrict__ out_w, const float* __restrict__ out_b,
                            float* __restrict__ logits) {
  __shared__ float qh[D_], h1[D_], sqr[K_], sqi[K_], nrm[D_], rr[D_], red[256];
  const int b = blockIdx.x;
  const int t = threadIdx.x;       // 0..255

  const int iq = idx[b * L_ + (L_ - 1)];
  qh[t] = embed[(size_t)iq * D_ + t];
  __syncthreads();

  float acc = qp_b1[t];
  for (int j = 0; j < D_; ++j) acc += qh[j] * qp_w1[j * D_ + t];
  h1[t] = tanhf(acc);
  __syncthreads();

  if (t < K_) {
    float p = qp_b2[t], a = qp_ab[t];
    for (int j = 0; j < D_; ++j) {
      p += h1[j] * qp_w2[j * K_ + t];
      a += qh[j] * qp_aw[j * K_ + t];
    }
    float phase = tanhf(p) * PI_F;
    float amp = softplus_f(a) + 0.1f;
    float s, c; sincosf(phase, &s, &c);
    sqr[t] = amp * c; sqi[t] = amp * s;
  }
  __syncthreads();

  float r = 0.f;
  for (int k = 0; k < K_; ++k) {
    r += mem_r[((size_t)b * K_ + k) * D_ + t] * sqr[k]
       + mem_i[((size_t)b * K_ + k) * D_ + t] * sqi[k];
  }
  r *= rsqrtf((float)(NP_ * K_));

  red[t] = r; __syncthreads();
  for (int s = 128; s > 0; s >>= 1) { if (t < s) red[t] += red[t + s]; __syncthreads(); }
  const float mu = red[0] / (float)D_;
  __syncthreads();
  const float dv = r - mu;
  red[t] = dv * dv; __syncthreads();
  for (int s = 128; s > 0; s >>= 1) { if (t < s) red[t] += red[t + s]; __syncthreads(); }
  const float var = red[0] / (float)D_;
  nrm[t] = dv * rsqrtf(var + 1e-5f) * ln_g[t] + ln_b[t];
  __syncthreads();

  float o = ob[t];
  for (int j = 0; j < D_; ++j) o += nrm[j] * ow[j * D_ + t];
  rr[t] = o;
  __syncthreads();

  for (int v = t; v < VOCAB_; v += 256) {
    float lg = out_b[v];
    for (int j = 0; j < D_; ++j) lg += rr[j] * out_w[j * VOCAB_ + v];
    logits[(size_t)b * VOCAB_ + v] = lg;
  }
}

// ---------------------------------------------------------------------------
// 7) output: zero-fill (B,L,VOCAB) with float4 stores, scatter logits at l=L-1
// ---------------------------------------------------------------------------
__global__ void outfill_kernel(const float* __restrict__ logits, float* __restrict__ out) {
  const size_t i4 = (size_t)blockIdx.x * blockDim.x + threadIdx.x;
  const size_t total4 = (size_t)B_ * L_ * VOCAB_ / 4;
  if (i4 >= total4) return;
  const size_t f = i4 * 4;
  const size_t per_b = (size_t)L_ * VOCAB_;
  const int b = (int)(f / per_b);
  const size_t rem = f % per_b;
  const int l = (int)(rem / VOCAB_);
  const int v = (int)(rem % VOCAB_);
  float4 val = make_float4(0.f, 0.f, 0.f, 0.f);
  if (l == L_ - 1) {
    const float* lp = logits + (size_t)b * VOCAB_ + v;
    val = make_float4(lp[0], lp[1], lp[2], lp[3]);
  }
  ((float4*)out)[i4] = val;
}

// ---------------------------------------------------------------------------
extern "C" void kernel_launch(void* const* d_in, const int* in_sizes, int n_in,
                              void* d_out, int out_size, void* d_ws, size_t ws_size,
                              hipStream_t stream) {
  const float* x      = (const float*)d_in[0];
  const float* embed  = (const float*)d_in[1];
  const float* kp_w1  = (const float*)d_in[2];
  const float* kp_b1  = (const float*)d_in[3];
  const float* kp_w2  = (const float*)d_in[4];
  const float* kp_b2  = (const float*)d_in[5];
  const float* kp_aw  = (const float*)d_in[6];
  const float* kp_ab  = (const float*)d_in[7];
  const float* qp_w1  = (const float*)d_in[8];
  const float* qp_b1  = (const float*)d_in[9];
  const float* qp_w2  = (const float*)d_in[10];
  const float* qp_b2  = (const float*)d_in[11];
  const float* qp_aw  = (const float*)d_in[12];
  const float* qp_ab  = (const float*)d_in[13];
  const float* vw     = (const float*)d_in[14];
  const float* vb     = (const float*)d_in[15];
  const float* ln_g   = (const float*)d_in[16];
  const float* ln_b   = (const float*)d_in[17];
  const float* ow     = (const float*)d_in[18];
  const float* ob     = (const float*)d_in[19];
  const float* out_w  = (const float*)d_in[20];
  const float* out_b  = (const float*)d_in[21];
  float* out = (float*)d_out;

  // workspace carve
  char* w = (char*)d_ws;
  auto alloc = [&](size_t bytes) -> void* {
    void* p = (void*)w;
    w += (bytes + 255) & ~(size_t)255;
    return p;
  };
  int*    idx      = (int*)   alloc((size_t)B_ * L_ * sizeof(int));
  bf16_t* keyh     = (bf16_t*)alloc((size_t)M_ * D_ * sizeof(bf16_t));
  bf16_t* valh     = (bf16_t*)alloc((size_t)M_ * D_ * sizeof(bf16_t));
  bf16_t* h1b      = (bf16_t*)alloc((size_t)M_ * D_ * sizeof(bf16_t));
  bf16_t* Vt       = (bf16_t*)alloc((size_t)B_ * D_ * NPP_ * sizeof(bf16_t));
  bf16_t* w1t      = (bf16_t*)alloc((size_t)D_ * D_ * sizeof(bf16_t));
  bf16_t* vwt      = (bf16_t*)alloc((size_t)D_ * D_ * sizeof(bf16_t));
  bf16_t* w2t      = (bf16_t*)alloc((size_t)D_ * K_ * sizeof(bf16_t));
  bf16_t* awt      = (bf16_t*)alloc((size_t)D_ * K_ * sizeof(bf16_t));
  float*  phasepre = (float*) alloc((size_t)M_ * K_ * sizeof(float));
  float*  amppre   = (float*) alloc((size_t)M_ * K_ * sizeof(float));
  bf16_t* krT      = (bf16_t*)alloc((size_t)B_ * K_ * NPP_ * sizeof(bf16_t));
  bf16_t* kiT      = (bf16_t*)alloc((size_t)B_ * K_ * NPP_ * sizeof(bf16_t));
  float*  mem_r    = (float*) alloc((size_t)B_ * K_ * D_ * sizeof(float));
  float*  mem_i    = (float*) alloc((size_t)B_ * K_ * D_ * sizeof(float));
  float*  logits   = (float*) alloc((size_t)B_ * VOCAB_ * sizeof(float));
  (void)ws_size; (void)n_in; (void)in_sizes; (void)out_size;

  // weight conversions (f32 [K][N] -> bf16 transposed [N][K])
  cvtT_kernel<<<(D_ * D_ + 255) / 256, 256, 0, stream>>>(kp_w1, w1t, D_, D_);
  cvtT_kernel<<<(D_ * D_ + 255) / 256, 256, 0, stream>>>(vw,    vwt, D_, D_);
  cvtT_kernel<<<(D_ * K_ + 255) / 256, 256, 0, stream>>>(kp_w2, w2t, D_, K_);
  cvtT_kernel<<<(D_ * K_ + 255) / 256, 256, 0, stream>>>(kp_aw, awt, D_, K_);

  // token indices + gather
  argmax_kernel<<<B_ * L_, 256, 0, stream>>>(x, idx);
  gather_kernel<<<M_, 256, 0, stream>>>(embed, idx, keyh, valh);

  // zero padded entries of krT/kiT/Vt (before producers fill real entries)
  padzero_kernel<<<(B_ * D_ + 255) / 256, 256, 0, stream>>>(krT, kiT, Vt);

  // big WMMA GEMMs over M = 8176 rows (grid.y = N/64, 4 waves/block)
  gemm_wmma_kernel<1, D_><<<dim3(M_ / 16, D_ / 64), 128, 0, stream>>>(keyh, w1t, kp_b1, (void*)h1b, D_);
  gemm_wmma_kernel<3, D_><<<dim3(M_ / 16, D_ / 64), 128, 0, stream>>>(valh, vwt, vb,    (void*)Vt,  D_);
  gemm_wmma_kernel<0, D_><<<dim3(M_ / 16, K_ / 64), 128, 0, stream>>>(h1b,  w2t, kp_b2, (void*)phasepre, K_);
  gemm_wmma_kernel<0, D_><<<dim3(M_ / 16, K_ / 64), 128, 0, stream>>>(keyh, awt, kp_ab, (void*)amppre,   K_);

  // kr/ki transcendentals -> transposed padded layout
  krki_kernel<<<(M_ * K_ + 255) / 256, 256, 0, stream>>>(phasepre, amppre, krT, kiT);

  // batched binding GEMM (mem_r, mem_i)
  bind_wmma_kernel<<<dim3(B_, K_ / 16, D_ / 64), 128, 0, stream>>>(krT, kiT, Vt, mem_r, mem_i);

  // head: query path + layernorm + logits
  head_kernel<<<B_, 256, 0, stream>>>(idx, embed, qp_w1, qp_b1, qp_w2, qp_b2, qp_aw, qp_ab,
                                      mem_r, mem_i, ln_g, ln_b, ow, ob, out_w, out_b, logits);

  // zero-fill output + scatter last-row logits
  const size_t total4 = (size_t)B_ * L_ * VOCAB_ / 4;
  outfill_kernel<<<(unsigned)((total4 + 255) / 256), 256, 0, stream>>>(logits, out);
}
